// Voltera_75428215652658
// MI455X (gfx1250) — matrix operand order repaired and checked by reading, not compiled, not run on hardware
//
#include <hip/hip_runtime.h>
#include <hip/hip_bf16.h>

// ---------------------------------------------------------------------------
// out[s,o] = sum_ij p01[s,ij] * ( sum_kl p23[s,kl] * w[ij*256+kl, o] )
//   p01 = c0 (x) c1, p23 = c2 (x) c3  (256 f32 each per sample)
// Inner contraction: bf16 WMMA GEMM; A (= p23 fragments) held in VGPRs,
// p01 applied as f32 scale in the epilogue. The ij sum is split 4-ways
// across blocks (split-K) for occupancy; partials reduced deterministically.
// ---------------------------------------------------------------------------

typedef __attribute__((ext_vector_type(16))) __bf16 v16bf;
typedef __attribute__((ext_vector_type(8)))  float  v8f;
typedef __attribute__((ext_vector_type(4)))  float  f32x4;

#define NSAMP 1024      // B*S
#define NOUT  64
#define NIJ   256
#define NSPLIT 4        // ij-split factor
#define IJQ   (NIJ / NSPLIT)   // 64 ij per block

// norm[n] = 1 / (2^((n-1)/2) * Gamma((n+1)/2)),  n = 1..15  (index 0 unused)
__constant__ float c_norm[16] = {
    0.0f,
    1.0f,            0.79788456080f,  0.5f,            0.26596152027f,
    0.125f,          0.05319230405f,  0.02083333333f,  0.00759891190f,
    0.00260416667f,  8.44322163e-4f,  2.60416667e-4f,  7.67571325e-5f,
    2.17013889e-5f,  5.90435797e-6f,  1.55009921e-6f };

// sig = 3.719 / sqrt(-2*ln(1-0.001)) = 83.138572 ;  inv2s2 = 0.5/sig^2
#define INV2S2 7.23377e-5f

// ---------------------------------------------------------------------------
// Kernel 1: per-sample polynomial factors.
//   p23  [s][kl]            (flat, row-major per sample)
//   p01t [s>>4][ij][s&15]   (transposed per 16-sample M tile -> contiguous
//                            scale vectors for the GEMM epilogue)
// grid = 1024 blocks x 256 threads
// ---------------------------------------------------------------------------
__global__ void voltera_poly_kernel(const float* __restrict__ x,
                                    float* __restrict__ p01t,
                                    float* __restrict__ p23) {
  __shared__ float cbuf[4][16];
  const int s = blockIdx.x;
  const int t = threadIdx.x;
  if (t < 64) {
    const int ch = t >> 4;
    const int ni = t & 15;
    const float xv = x[s * 4 + ch];
    float val = 1.0f;
    if (ni > 0) {
      float pw = xv;
      for (int q = 1; q < ni; ++q) pw *= xv;       // x^ni (exact for int exp)
      const float win = __expf(-xv * xv * INV2S2);
      val = pw * win * c_norm[ni];
    }
    cbuf[ch][ni] = val;
  }
  __syncthreads();
  const int i = t >> 4, j = t & 15;
  p23[(size_t)s * 256 + t] = cbuf[2][i] * cbuf[3][j];
  p01t[(size_t)(s >> 4) * 4096 + t * 16 + (s & 15)] = cbuf[0][i] * cbuf[1][j];
}

// ---------------------------------------------------------------------------
// Kernel 2: convert w (f32, K x 64) -> bf16, pre-swizzled into WMMA B-operand
// lane order. Fragment (kc, nt): lane L holds column n = nt*16 + (L&15),
// K rows kc*32 + (L>>4)*8 + (e&7) + (e>>3)*16, e = 0..15 (per-lane 16 bf16).
// grid = 1024 blocks x 256 threads (8192 fragments * 32 lanes)
// ---------------------------------------------------------------------------
__global__ void voltera_wswz_kernel(const float* __restrict__ w,
                                    __bf16* __restrict__ wswz) {
  const int gt   = blockIdx.x * 256 + threadIdx.x;
  const int lane = gt & 31;
  const int frag = gt >> 5;          // 0..8191 = kc*4 + nt
  const int kc = frag >> 2, nt = frag & 3;
  const int n  = nt * 16 + (lane & 15);
  const int kb = kc * 32 + (lane >> 4) * 8;
  v16bf v;
#pragma unroll
  for (int e = 0; e < 16; ++e)
    v[e] = (__bf16)w[(size_t)(kb + (e & 7) + (e >> 3) * 16) * NOUT + n];
  *(v16bf*)(wswz + (size_t)frag * 512 + lane * 16) = v;
}

// ---------------------------------------------------------------------------
// Kernel 3: GEMM, split-K over ij. 256 blocks x 128 threads (4 waves);
//   mt = blockIdx.x>>2 (M tile), ijc = blockIdx.x&3 (ij quarter),
//   wave nt owns output columns nt*16..nt*16+15.
// A fragments (p23 in bf16) built ONCE and held in 64 VGPRs; per-ij inner
// loop is 8 WMMAs (two independent chains) + B loads + f32 scale/accumulate.
// Each quarter writes its own partial buffer (deterministic reduction after).
// ---------------------------------------------------------------------------
__global__ void __launch_bounds__(128)
voltera_gemm_kernel(const float* __restrict__ p01t,
                    const float* __restrict__ p23,
                    const __bf16* __restrict__ wswz,
                    float* __restrict__ part) {
  __shared__ float lp23[16 * 256];    // 16 KB
  __shared__ float lp01q[IJQ * 16];   //  4 KB (owned ij quarter)

  const int mt  = blockIdx.x >> 2;    // M tile
  const int ijc = blockIdx.x & 3;     // ij quarter
  const int s0  = mt * 16;
  const int t   = threadIdx.x;
  const int nt   = t >> 5;            // wave id == N tile
  const int lane = t & 31;

  // cooperative tile loads
  {
    const f32x4* g23 = (const f32x4*)(p23  + (size_t)s0 * 256);
    const f32x4* g01 = (const f32x4*)(p01t + (size_t)mt * 4096 + ijc * (IJQ * 16));
    f32x4* l23 = (f32x4*)lp23;
    f32x4* l01 = (f32x4*)lp01q;
#pragma unroll
    for (int q = 0; q < 8; ++q) l23[t + q * 128] = g23[t + q * 128];  // 4096 f32
#pragma unroll
    for (int q = 0; q < 2; ++q) l01[t + q * 128] = g01[t + q * 128];  // 1024 f32
  }
  __syncthreads();

  const int m = lane & 15;
  const int h = lane >> 4;

  // Build 8 A fragments (pure p23, bf16) once.
  // 16-bit A 16x32 layout: lane L -> row m = L&15, half h = L>>4;
  // K offsets {8h..8h+7, 16+8h..16+8h+7} within the chunk.
  v16bf afrag[8];
  {
    const f32x4* l23v = (const f32x4*)lp23;
    const int rowbase = m * 64;       // m*256 floats, in float4 units
#pragma unroll
    for (int c = 0; c < 8; ++c) {
      const int b0 = rowbase + ((c * 32 + 8 * h) >> 2);
      const f32x4 r0 = l23v[b0];
      const f32x4 r1 = l23v[b0 + 1];
      const f32x4 r2 = l23v[b0 + 4];
      const f32x4 r3 = l23v[b0 + 5];
#pragma unroll
      for (int e = 0; e < 4; ++e) {
        afrag[c][e]      = (__bf16)r0[e];
        afrag[c][4 + e]  = (__bf16)r1[e];
        afrag[c][8 + e]  = (__bf16)r2[e];
        afrag[c][12 + e] = (__bf16)r3[e];
      }
    }
  }

  const f32x4* l01v = (const f32x4*)lp01q;
  v8f oacc = {};

#pragma unroll 1
  for (int ijq = 0; ijq < IJQ; ++ijq) {
    const int ij = ijc * IJQ + ijq;
    // B fragments for this ij: global K chunks kc = ij*8 + c, c = 0..7
    const __bf16* bbase = wswz + ((size_t)(ij * 8) * 4 + nt) * 512 + lane * 16;

    v8f acc0 = {}, acc1 = {};
#pragma unroll
    for (int c = 0; c < 8; c += 2) {
      const v16bf b0 = *(const v16bf*)(bbase + (size_t)c * 2048);
      const v16bf b1 = *(const v16bf*)(bbase + (size_t)(c + 1) * 2048);
      acc0 = __builtin_amdgcn_wmma_f32_16x16x32_bf16(
          false, afrag[c],     false, b0, (short)0, acc0, false, false);
      acc1 = __builtin_amdgcn_wmma_f32_16x16x32_bf16(
          false, afrag[c + 1], false, b1, (short)0, acc1, false, false);
    }

    // prefetch next ij's B stream (global_prefetch_b8)
    __builtin_prefetch(bbase + (size_t)8 * 2048, 0, 1);

    // f32 epilogue: oacc[r] += p01[row r+8h, ij] * (acc0[r] + acc1[r])
    // scales contiguous in lp01q: lp01q[ijq*16 + 8h .. +7]
    const f32x4 sc0 = l01v[ijq * 4 + 2 * h];
    const f32x4 sc1 = l01v[ijq * 4 + 2 * h + 1];
#pragma unroll
    for (int r = 0; r < 4; ++r) {
      oacc[r]     += sc0[r] * (acc0[r]     + acc1[r]);
      oacc[4 + r] += sc1[r] * (acc0[4 + r] + acc1[4 + r]);
    }
  }

  // D layout: lane L -> column n = L&15; VGPR r -> row m = r + 8*(L>>4)
  float* pout = part + (size_t)ijc * (NSAMP * NOUT);
  const int ncol = lane & 15;
#pragma unroll
  for (int r = 0; r < 8; ++r) {
    pout[(size_t)(s0 + r + 8 * h) * NOUT + nt * 16 + ncol] = oacc[r];
  }
}

// ---------------------------------------------------------------------------
// Kernel 4: deterministic fixed-order reduction of the 4 split-K partials.
// grid = 64 blocks x 256 threads, float4 per thread (65536 f32 total)
// ---------------------------------------------------------------------------
__global__ void voltera_reduce_kernel(const float* __restrict__ part,
                                      float* __restrict__ out) {
  const int i = blockIdx.x * 256 + threadIdx.x;   // float4 index, 0..16383
  const f32x4* p = (const f32x4*)part;
  f32x4 a = p[i];
  a = a + p[i + 16384];
  a = a + p[i + 2 * 16384];
  a = a + p[i + 3 * 16384];
  ((f32x4*)out)[i] = a;
}

// ---------------------------------------------------------------------------
// Workspace layout (bytes):
//   [0, 1MB)        p01t : 64 tiles * 256 ij * 16 rows f32
//   [1MB, 2MB)      p23  : 1024*256 f32
//   [2MB, 3MB)      part : 4 * 1024 * 64 f32 (split-K partials)
//   [3MB, ~11.4MB)  wswz : 2048*4*512 bf16 (8 MB)
// ---------------------------------------------------------------------------
extern "C" void kernel_launch(void* const* d_in, const int* in_sizes, int n_in,
                              void* d_out, int out_size, void* d_ws, size_t ws_size,
                              hipStream_t stream) {
  const float* x = (const float*)d_in[0];   // (2, 512, 4) f32
  const float* w = (const float*)d_in[1];   // (65536, 64) f32
  float* out = (float*)d_out;               // (2, 512, 64) f32

  char* ws = (char*)d_ws;
  float*  p01t = (float*)(ws);
  float*  p23  = (float*)(ws + (size_t)1 * 1024 * 1024);
  float*  part = (float*)(ws + (size_t)2 * 1024 * 1024);
  __bf16* wswz = (__bf16*)(ws + (size_t)3 * 1024 * 1024);

  voltera_poly_kernel<<<NSAMP, 256, 0, stream>>>(x, p01t, p23);
  voltera_wswz_kernel<<<1024, 256, 0, stream>>>(w, wswz);
  voltera_gemm_kernel<<<(NSAMP / 16) * NSPLIT, 128, 0, stream>>>(p01t, p23, wswz, part);
  voltera_reduce_kernel<<<64, 256, 0, stream>>>(part, out);
}
